// GolayLinear_21303037788552
// MI455X (gfx1250) — matrix operand order repaired and checked by reading, compiled, not run on hardware
//
#include <hip/hip_runtime.h>
#include <cstdint>

#define IN_F   4096
#define OUT_F  4096
#define MROWS  8192   // 4 * 2048

typedef __attribute__((ext_vector_type(16))) __bf16 v16bf;
typedef __attribute__((ext_vector_type(8)))  float  v8f;

union Frag32B {
    struct { uint4 a, b; } u;   // 32 bytes = 16 bf16
    v16bf v;
};

__device__ __forceinline__ unsigned bf16_rne_bits(float f) {
    unsigned u = __float_as_uint(f);
    return (u + 0x7FFFu + ((u >> 16) & 1u)) >> 16;
}

// Split f32 pair into packed bf16 hi (truncate) and bf16 lo (RNE of residual).
__device__ __forceinline__ void split_pack2(float e0, float e1,
                                            unsigned& hi, unsigned& lo) {
    unsigned u0 = __float_as_uint(e0), u1 = __float_as_uint(e1);
    hi = (u0 >> 16) | (u1 & 0xFFFF0000u);
    float r0 = e0 - __uint_as_float(u0 & 0xFFFF0000u);
    float r1 = e1 - __uint_as_float(u1 & 0xFFFF0000u);
    lo = bf16_rne_bits(r0) | (bf16_rne_bits(r1) << 16);
}

// Async global->LDS 16B per lane (ASYNCcnt tracked).
// NOTE: INST_OFFSET is applied to BOTH the LDS and global addresses.
#define ASYNC_B128(ldsaddr, gaddr, OFFS)                                     \
    asm volatile("global_load_async_to_lds_b128 %0, %1, off offset:" #OFFS  \
                 :: "v"(ldsaddr), "v"(gaddr) : "memory")

#define WAIT_ASYNC0() asm volatile("s_wait_asynccnt 0" ::: "memory")

// ---------------------------------------------------------------------------
// Kernel 1: fold FWHT + golay into W.  w2[o,d] = golay[d] * fwht(W[o,:])[d],
// split into bf16 hi/lo planes.  One block per output row, 16KB LDS.
// ---------------------------------------------------------------------------
__global__ __launch_bounds__(256)
void fwht_w_kernel(const float* __restrict__ W,
                   const float* __restrict__ golay,
                   unsigned short* __restrict__ w2h,
                   unsigned short* __restrict__ w2l) {
    __shared__ float s[IN_F];
    const int o = blockIdx.x;
    const int t = threadIdx.x;
    const float* wrow = W + (size_t)o * IN_F;

    #pragma unroll
    for (int i = 0; i < IN_F / 256; ++i) s[t + 256 * i] = wrow[t + 256 * i];
    __syncthreads();

    const float is2 = 0.70710678118654752440f;
    for (int h = 1; h < IN_F; h <<= 1) {
        #pragma unroll
        for (int i = 0; i < (IN_F / 2) / 256; ++i) {
            int p  = t + 256 * i;
            int i0 = ((p & ~(h - 1)) << 1) | (p & (h - 1));
            int i1 = i0 + h;
            float a  = s[i0];
            float bb = s[i1];
            s[i0] = (a + bb) * is2;
            s[i1] = (a - bb) * is2;
        }
        __syncthreads();
    }

    #pragma unroll
    for (int i = 0; i < IN_F / 256; ++i) {
        int   d = t + 256 * i;
        float w = s[d] * golay[d];
        unsigned u  = __float_as_uint(w);
        float    hi = __uint_as_float(u & 0xFFFF0000u);
        w2h[(size_t)o * IN_F + d] = (unsigned short)(u >> 16);
        w2l[(size_t)o * IN_F + d] = (unsigned short)bf16_rne_bits(w - hi);
    }
}

// ---------------------------------------------------------------------------
// Kernel 2: out = x @ w2^T + b, 3-term bf16 WMMA (f32-accurate).
// 128(M) x 256(N) block tile, 8 waves (2x4), 64x64 per wave, K-step 32.
// Double-buffered LDS arena (integer-offset buffer select keeps addrspace(3));
// B planes fetched async global->LDS; one barrier per K-step.
// ---------------------------------------------------------------------------

// Arena offsets in uint4 (16B) units.  Rows padded to 80B (5 uint4).
#define A_U4      (128 * 5)             // 640  : one A plane
#define B_U4      (256 * 5)             // 1280 : one B plane
#define U4_AH0    0
#define U4_AL0    (2 * A_U4)            // 1280
#define U4_BH0    (4 * A_U4)            // 2560
#define U4_BL0    (U4_BH0 + 2 * B_U4)   // 5120
#define SMEM_U4   (U4_BL0 + 2 * B_U4)   // 7680  => 120KB

__global__ __launch_bounds__(256)
void golay_gemm_kernel(const float* __restrict__ X,
                       const unsigned short* __restrict__ w2h,
                       const unsigned short* __restrict__ w2l,
                       const float* __restrict__ bias,
                       float* __restrict__ out) {
    __shared__ uint4 smem4[SMEM_U4];

    const int tid  = threadIdx.x;
    const int lane = tid & 31;
    const int wave = tid >> 5;
    const int lm   = lane & 15;
    const int lh   = lane >> 4;
    const int wr   = wave >> 2;      // 0..1
    const int wc   = wave & 3;       // 0..3

    const int m0 = blockIdx.y * 128;
    const int n0 = blockIdx.x * 256;

    // Raw LDS byte addresses for async destinations (low 32 bits of the flat
    // address inside the LDS aperture are the LDS byte offset).
    const unsigned lbase = (unsigned)(uintptr_t)(void*)smem4;
    const unsigned bhA[2] = { lbase + U4_BH0 * 16u          + (unsigned)tid * 80u,
                              lbase + (U4_BH0 + B_U4) * 16u + (unsigned)tid * 80u };
    const unsigned blA[2] = { lbase + U4_BL0 * 16u          + (unsigned)tid * 80u,
                              lbase + (U4_BL0 + B_U4) * 16u + (unsigned)tid * 80u };

    // A staging: 2 threads per row, 16 f32 each
    const int am   = tid >> 1;
    const int ak16 = (tid & 1);
    const float4* xrow = (const float4*)(X + (size_t)(m0 + am) * IN_F) + ak16 * 4;
    const int aIdx = am * 5 + ak16 * 2;

    // B staging: 1 thread per row, 64B per plane per K-step, async to LDS
    const char* gBh = (const char*)(w2h + (size_t)(n0 + tid) * IN_F);
    const char* gBl = (const char*)(w2l + (size_t)(n0 + tid) * IN_F);

    v8f acc[4][4];
    v8f zero = {0.f, 0.f, 0.f, 0.f, 0.f, 0.f, 0.f, 0.f};
    #pragma unroll
    for (int i = 0; i < 4; ++i)
        #pragma unroll
        for (int j = 0; j < 4; ++j) acc[i][j] = zero;

    // ---- prologue: stage block 0 into buffer 0 ----
    {
        float4 f0 = xrow[0], f1 = xrow[1], f2 = xrow[2], f3 = xrow[3];
        xrow += 8;
        uint4 h0, l0, h1, l1;
        split_pack2(f0.x, f0.y, h0.x, l0.x);  split_pack2(f0.z, f0.w, h0.y, l0.y);
        split_pack2(f1.x, f1.y, h0.z, l0.z);  split_pack2(f1.z, f1.w, h0.w, l0.w);
        split_pack2(f2.x, f2.y, h1.x, l1.x);  split_pack2(f2.z, f2.w, h1.y, l1.y);
        split_pack2(f3.x, f3.y, h1.z, l1.z);  split_pack2(f3.z, f3.w, h1.w, l1.w);
        smem4[U4_AH0 + aIdx] = h0;  smem4[U4_AH0 + aIdx + 1] = h1;
        smem4[U4_AL0 + aIdx] = l0;  smem4[U4_AL0 + aIdx + 1] = l1;

        ASYNC_B128(bhA[0], gBh, 0);  ASYNC_B128(bhA[0], gBh, 16);
        ASYNC_B128(bhA[0], gBh, 32); ASYNC_B128(bhA[0], gBh, 48);
        ASYNC_B128(blA[0], gBl, 0);  ASYNC_B128(blA[0], gBl, 16);
        ASYNC_B128(blA[0], gBl, 32); ASYNC_B128(blA[0], gBl, 48);
        gBh += 64; gBl += 64;
    }

    int cur = 0;
    #pragma unroll 2
    for (int k0 = 0; k0 < IN_F; k0 += 32) {
        WAIT_ASYNC0();          // B[cur] fully landed in LDS
        __syncthreads();        // A[cur] stores + everyone's B wait done

        const int nxt = cur ^ 1;
        if (k0 + 32 < IN_F) {
            // async B copy for next K-block overlaps this block's WMMAs
            ASYNC_B128(bhA[nxt], gBh, 0);  ASYNC_B128(bhA[nxt], gBh, 16);
            ASYNC_B128(bhA[nxt], gBh, 32); ASYNC_B128(bhA[nxt], gBh, 48);
            ASYNC_B128(blA[nxt], gBl, 0);  ASYNC_B128(blA[nxt], gBl, 16);
            ASYNC_B128(blA[nxt], gBl, 32); ASYNC_B128(blA[nxt], gBl, 48);
            gBh += 64; gBl += 64;

            // stage A for next K-block (f32 -> bf16 hi/lo split)
            float4 f0 = xrow[0], f1 = xrow[1], f2 = xrow[2], f3 = xrow[3];
            xrow += 8;
            __builtin_prefetch(xrow, 0, 1);   // x tile two K-steps ahead
            uint4 h0, l0, h1, l1;
            split_pack2(f0.x, f0.y, h0.x, l0.x);  split_pack2(f0.z, f0.w, h0.y, l0.y);
            split_pack2(f1.x, f1.y, h0.z, l0.z);  split_pack2(f1.z, f1.w, h0.w, l0.w);
            split_pack2(f2.x, f2.y, h1.x, l1.x);  split_pack2(f2.z, f2.w, h1.y, l1.y);
            split_pack2(f3.x, f3.y, h1.z, l1.z);  split_pack2(f3.z, f3.w, h1.w, l1.w);
            const int aH = U4_AH0 + nxt * A_U4;
            const int aL = U4_AL0 + nxt * A_U4;
            smem4[aH + aIdx] = h0;  smem4[aH + aIdx + 1] = h1;
            smem4[aL + aIdx] = l0;  smem4[aL + aIdx + 1] = l1;
        }

        // ---- compute: 48 v_wmma_f32_16x16x32_bf16 per wave per K-step ----
        const int cAh = U4_AH0 + cur * A_U4;
        const int cAl = U4_AL0 + cur * A_U4;
        const int cBh = U4_BH0 + cur * B_U4;
        const int cBl = U4_BL0 + cur * B_U4;

        Frag32B Ah[4], Al[4];
        #pragma unroll
        for (int im = 0; im < 4; ++im) {
            const int arow = wr * 64 + im * 16 + lm;
            const int i1 = arow * 5 + lh;
            Ah[im].u.a = smem4[cAh + i1];  Ah[im].u.b = smem4[cAh + i1 + 2];
            Al[im].u.a = smem4[cAl + i1];  Al[im].u.b = smem4[cAl + i1 + 2];
        }
        #pragma unroll
        for (int jn = 0; jn < 4; ++jn) {
            const int brow = wc * 64 + jn * 16 + lm;
            const int i1 = brow * 5 + lh;
            Frag32B Bh, Bl;
            Bh.u.a = smem4[cBh + i1];  Bh.u.b = smem4[cBh + i1 + 2];
            Bl.u.a = smem4[cBl + i1];  Bl.u.b = smem4[cBl + i1 + 2];
            #pragma unroll
            for (int im = 0; im < 4; ++im) {
                acc[im][jn] = __builtin_amdgcn_wmma_f32_16x16x32_bf16(
                    false, Ah[im].v, false, Bh.v, (short)0, acc[im][jn], false, false);
                acc[im][jn] = __builtin_amdgcn_wmma_f32_16x16x32_bf16(
                    false, Ah[im].v, false, Bl.v, (short)0, acc[im][jn], false, false);
                acc[im][jn] = __builtin_amdgcn_wmma_f32_16x16x32_bf16(
                    false, Al[im].v, false, Bh.v, (short)0, acc[im][jn], false, false);
            }
        }
        cur = nxt;
    }

    // ---- epilogue: C/D layout is VGPR v -> M = v + 8*lh, N = lm ----
    #pragma unroll
    for (int jn = 0; jn < 4; ++jn) {
        const int col = n0 + wc * 64 + jn * 16 + lm;
        const float bv = bias[col];
        #pragma unroll
        for (int im = 0; im < 4; ++im) {
            const int mb = m0 + wr * 64 + im * 16 + lh * 8;
            float* op = out + (size_t)mb * OUT_F + col;
            #pragma unroll
            for (int v = 0; v < 8; ++v)
                op[(size_t)v * OUT_F] = acc[im][jn][v] + bv;
        }
    }
}

// ---------------------------------------------------------------------------
extern "C" void kernel_launch(void* const* d_in, const int* in_sizes, int n_in,
                              void* d_out, int out_size, void* d_ws, size_t ws_size,
                              hipStream_t stream) {
    const float* x     = (const float*)d_in[0];   // (4,2048,4096) f32
    const float* golay = (const float*)d_in[1];   // (4096,)       f32
    const float* W     = (const float*)d_in[2];   // (4096,4096)   f32
    const float* b     = (const float*)d_in[3];   // (4096,)       f32
    float*       out   = (float*)d_out;           // (4,2048,4096) f32

    unsigned short* w2h = (unsigned short*)d_ws;
    unsigned short* w2l = w2h + (size_t)OUT_F * IN_F;

    fwht_w_kernel<<<OUT_F, 256, 0, stream>>>(W, golay, w2h, w2l);

    dim3 grid(OUT_F / 256, MROWS / 128);
    golay_gemm_kernel<<<grid, 256, 0, stream>>>(x, w2h, w2l, b, out);
}
